// Transformer_7121055777383
// MI455X (gfx1250) — matrix operand order repaired and checked by reading, not compiled, not run on hardware
//
#include <hip/hip_runtime.h>
#include <hip/hip_bf16.h>
#include <stdint.h>

// ---------------------------------------------------------------------------
// MI455X (gfx1250) transformer forward: bf16 WMMA GEMMs + f32 reductions.
// Async global->LDS staging (GLOBAL_LOAD_ASYNC_TO_LDS_B128) with LDS double
// buffering so the Tensor-less DMA of tile t+1 overlaps WMMA on tile t.
// ---------------------------------------------------------------------------
#define TDIM   512
#define DDIM   1024
#define HHEADS 16
#define HD     64
#define FFDIM  4096
#define BB     8
#define NTOK   4096          // BB * TDIM
#define VY     32000

typedef __bf16 bf16;
typedef __attribute__((ext_vector_type(16))) __bf16   v16bf;
typedef __attribute__((ext_vector_type(8)))  float    v8f;
typedef __attribute__((ext_vector_type(4)))  unsigned u32x4;

union ABfrag { v16bf v; u32x4 u[2]; };

__device__ __forceinline__ bf16 f2bf(float f) {
  unsigned u = __builtin_bit_cast(unsigned, f);
  unsigned r = u + 0x7FFFu + ((u >> 16) & 1u);     // round-to-nearest-even
  unsigned short h = (unsigned short)(r >> 16);
  return __builtin_bit_cast(bf16, h);
}

// ---------------------------------------------------------------------------
// WMMA GEMM:  C[M,N] = A[M,K] (bf16, row-major, lda=K)
//                    x Bt[N,K] (bf16, row-major = B^T, ldb=K)
// 256 threads = 8 wave32 waves. Each wave computes WMxWN via 16x16x32 WMMAs.
// K-tiles staged into double-buffered LDS with async-to-LDS DMA (ASYNCcnt).
// EPI 0: +bias[n] (opt ReLU), f32 out (+ optional bf16 mirror)
// EPI 1: attention scores: *scale, optional causal mask, per-z [512,512] out
// EPI 2: att@V head-merge store into [B,T,D] (f32 + bf16 mirror)
// ---------------------------------------------------------------------------
template<int BM, int BN, int WM, int WN, int EPI, bool RELU>
__global__ __launch_bounds__(256)
void gemm_kernel(const bf16* __restrict__ A,  int lda, long long zsA,
                 const bf16* __restrict__ Bt, int ldb, long long zsB,
                 float* __restrict__ C, int ldc,
                 const float* __restrict__ bias,
                 bf16* __restrict__ Cbf,
                 float scale, int causal, int K)
{
  constexpr int BK = 32, PAD = 8, LDT = BK + PAD;   // rows stay 16B aligned
  __shared__ __align__(16) bf16 As[2][BM * LDT];
  __shared__ __align__(16) bf16 Bs[2][BN * LDT];

  const int tid  = threadIdx.x;
  const int lane = tid & 31;
  const int wave = tid >> 5;
  const int z    = blockIdx.z;

  const bf16* Ag = A  + (long long)z * zsA + (size_t)blockIdx.x * BM * lda;
  const bf16* Bg = Bt + (long long)z * zsB + (size_t)blockIdx.y * BN * ldb;

  // ---- per-thread async copy slots (16B chunks), fully unrolled -----------
  constexpr int NCA = (BM * (BK / 8)) / 256;   // chunks per thread for A tile
  constexpr int NCB = (BN * (BK / 8)) / 256;   // chunks per thread for B tile
  constexpr unsigned bufA = (unsigned)(BM * LDT * sizeof(bf16));
  constexpr unsigned bufB = (unsigned)(BN * LDT * sizeof(bf16));

  unsigned aLds[NCA]; unsigned long long aGa[NCA];
#pragma unroll
  for (int i = 0; i < NCA; ++i) {
    int c = tid + i * 256, r = c >> 2, kc = (c & 3) * 8;
    aLds[i] = (unsigned)(uintptr_t)&As[0][r * LDT + kc];     // LDS byte addr
    aGa[i]  = (unsigned long long)(uintptr_t)(Ag + (size_t)r * lda + kc);
  }
  unsigned bLds[NCB]; unsigned long long bGa[NCB];
#pragma unroll
  for (int i = 0; i < NCB; ++i) {
    int c = tid + i * 256, r = c >> 2, kc = (c & 3) * 8;
    bLds[i] = (unsigned)(uintptr_t)&Bs[0][r * LDT + kc];
    bGa[i]  = (unsigned long long)(uintptr_t)(Bg + (size_t)r * ldb + kc);
  }

  // async-stage one (A,B) K-tile into LDS buffer `buf`
  auto stage = [&](int buf, int k0) {
    const unsigned long long koff = (unsigned long long)k0 * sizeof(bf16);
    const unsigned la = buf ? bufA : 0u;
    const unsigned lb = buf ? bufB : 0u;
#pragma unroll
    for (int i = 0; i < NCA; ++i)
      asm volatile("global_load_async_to_lds_b128 %0, %1, off"
                   :: "v"(aLds[i] + la), "v"(aGa[i] + koff) : "memory");
#pragma unroll
    for (int i = 0; i < NCB; ++i)
      asm volatile("global_load_async_to_lds_b128 %0, %1, off"
                   :: "v"(bLds[i] + lb), "v"(bGa[i] + koff) : "memory");
  };

  constexpr int WNV = BN / WN;
  const int waveM = wave / WNV;
  const int waveN = wave % WNV;
  constexpr int IM = WM / 16, IN = WN / 16;

  const v8f vzero = {0.f,0.f,0.f,0.f,0.f,0.f,0.f,0.f};
  v8f acc[IM][IN];
#pragma unroll
  for (int i = 0; i < IM; ++i)
#pragma unroll
    for (int j = 0; j < IN; ++j) acc[i][j] = vzero;

  const int hsel = lane >> 4;    // 0: lanes 0-15, 1: lanes 16-31
  const int l15  = lane & 15;

  // ---- software pipeline: prefetch tile 0, then overlap DMA with WMMA -----
  stage(0, 0);
  asm volatile("s_wait_asynccnt 0" ::: "memory");
  __syncthreads();

  const int nk = K / BK;
  for (int kt = 0; kt < nk; ++kt) {
    const int cur = kt & 1;
    if (kt + 1 < nk) stage(cur ^ 1, (kt + 1) * BK);   // DMA next tile

    // fragments per documented CDNA5 16-bit A/B VGPR layouts
    ABfrag fa[IM], fb[IN];
#pragma unroll
    for (int i = 0; i < IM; ++i) {
      const bf16* p = &As[cur][(waveM * WM + i * 16 + l15) * LDT + hsel * 8];
      fa[i].u[0] = *(const u32x4*)p;           // K = kb .. kb+7
      fa[i].u[1] = *(const u32x4*)(p + 16);    // K = 16+kb .. 16+kb+7
    }
#pragma unroll
    for (int j = 0; j < IN; ++j) {
      const bf16* p = &Bs[cur][(waveN * WN + j * 16 + l15) * LDT + hsel * 8];
      fb[j].u[0] = *(const u32x4*)p;           // K = kb .. kb+7
      fb[j].u[1] = *(const u32x4*)(p + 16);    // K = 16+kb .. 16+kb+7
    }
#pragma unroll
    for (int i = 0; i < IM; ++i)
#pragma unroll
      for (int j = 0; j < IN; ++j)
        acc[i][j] = __builtin_amdgcn_wmma_f32_16x16x32_bf16(
            false, fa[i].v, false, fb[j].v, (short)0, acc[i][j], false, false);

    // drain this wave's async DMA, then make the new tile visible to all
    asm volatile("s_wait_asynccnt 0" ::: "memory");
    __syncthreads();
  }

  // ---- epilogue (C/D layout: VGPR r -> M = r + 8*hsel, N = lane&15) -------
#pragma unroll
  for (int i = 0; i < IM; ++i) {
#pragma unroll
    for (int j = 0; j < IN; ++j) {
      const int nloc  = waveN * WN + j * 16 + l15;
      const int mbase = waveM * WM + i * 16 + hsel * 8;
      const int gn = blockIdx.y * BN + nloc;
#pragma unroll
      for (int r = 0; r < 8; ++r) {
        const int gm = blockIdx.x * BM + mbase + r;
        float v = acc[i][j][r];
        if constexpr (EPI == 0) {
          if (bias) v += bias[gn];
          if constexpr (RELU) v = fmaxf(v, 0.f);
          C[(size_t)gm * ldc + gn] = v;
          if (Cbf) Cbf[(size_t)gm * ldc + gn] = f2bf(v);
        } else if constexpr (EPI == 1) {
          v *= scale;
          if (causal && gn > gm) v = -1e10f;
          (C + (size_t)z * TDIM * TDIM)[(size_t)gm * TDIM + gn] = v;
        } else { // EPI == 2: z = b*H + h ; store at out[b, gm, h*64+gn]
          float* Cp = C   + (size_t)(z >> 4) * TDIM * DDIM + (size_t)(z & 15) * HD;
          bf16*  Cb = Cbf + (size_t)(z >> 4) * TDIM * DDIM + (size_t)(z & 15) * HD;
          Cp[(size_t)gm * DDIM + gn] = v;
          Cb[(size_t)gm * DDIM + gn] = f2bf(v);
        }
      }
    }
  }
}

// --------------------------- helper kernels --------------------------------

// W [K,N] f32 -> Wt [N,K] bf16
__global__ __launch_bounds__(256)
void wt_kernel(const float* __restrict__ W, bf16* __restrict__ Wt, int Kd, int Nd) {
  size_t i = (size_t)blockIdx.x * 256 + threadIdx.x;
  size_t total = (size_t)Kd * Nd;
  if (i >= total) return;
  int k = (int)(i / Nd), n = (int)(i % Nd);
  Wt[(size_t)n * Kd + k] = f2bf(W[i]);
}

// x[b,t,:] = emb[tok[b,t]] * sqrt(D) + pos[t]
__global__ __launch_bounds__(256)
void embed_kernel(const int* __restrict__ tok, const float* __restrict__ emb,
                  const float* __restrict__ pos,
                  float* __restrict__ out, bf16* __restrict__ outbf) {
  size_t i = (size_t)blockIdx.x * 256 + threadIdx.x;   // over NTOK*DDIM
  int d  = (int)(i & (DDIM - 1));
  int tg = (int)(i >> 10);
  int t  = tg & (TDIM - 1);
  int token = tok[tg];
  float v = emb[(size_t)token * DDIM + d] * 32.0f + pos[(size_t)t * DDIM + d];
  out[i] = v;
  outbf[i] = f2bf(v);
}

// [B,T,D] f32 -> bf16 [B,H,T,64] (outh) and/or [B,H,64,T] (outvt)
__global__ __launch_bounds__(256)
void headsplit_kernel(const float* __restrict__ in,
                      bf16* __restrict__ outh, bf16* __restrict__ outvt) {
  size_t i = (size_t)blockIdx.x * 256 + threadIdx.x;   // over BB*TDIM*DDIM
  int c = (int)(i & (DDIM - 1));
  int t = (int)((i >> 10) & (TDIM - 1));
  int b = (int)(i >> 19);
  int h = c >> 6, d = c & 63;
  bf16 v = f2bf(in[i]);
  size_t zh = (size_t)(b * HHEADS + h);
  if (outh)  outh [(zh * TDIM + t) * HD + d] = v;
  if (outvt) outvt[(zh * HD + d) * TDIM + t] = v;
}

// out = LN(x + res) * g + b   (res may be null); one block per row, D=1024
__global__ __launch_bounds__(256)
void ln_kernel(const float* __restrict__ x, const float* __restrict__ res,
               const float* __restrict__ g, const float* __restrict__ bta,
               float* __restrict__ out, bf16* __restrict__ outbf) {
  const int row = blockIdx.x, tid = threadIdx.x;
  const float* xr = x + (size_t)row * DDIM;
  const float* rr = res ? res + (size_t)row * DDIM : nullptr;
  float v[4]; float s = 0.f;
#pragma unroll
  for (int j = 0; j < 4; ++j) {
    int c = tid + j * 256;
    v[j] = xr[c] + (rr ? rr[c] : 0.f);
    s += v[j];
  }
  __shared__ float red[256];
  red[tid] = s; __syncthreads();
  for (int o = 128; o > 0; o >>= 1) { if (tid < o) red[tid] += red[tid + o]; __syncthreads(); }
  float mu = red[0] * (1.f / DDIM);
  __syncthreads();
  float s2 = 0.f;
#pragma unroll
  for (int j = 0; j < 4; ++j) { float d = v[j] - mu; s2 += d * d; }
  red[tid] = s2; __syncthreads();
  for (int o = 128; o > 0; o >>= 1) { if (tid < o) red[tid] += red[tid + o]; __syncthreads(); }
  float rs = rsqrtf(red[0] * (1.f / DDIM) + 1e-5f);
#pragma unroll
  for (int j = 0; j < 4; ++j) {
    int c = tid + j * 256;
    float o2 = (v[j] - mu) * rs * g[c] + bta[c];
    out  [(size_t)row * DDIM + c] = o2;
    outbf[(size_t)row * DDIM + c] = f2bf(o2);
  }
}

// row softmax over 512 cols; writes f32 (possibly in-place / into d_out) + bf16
__global__ __launch_bounds__(256)
void softmax_kernel(const float* __restrict__ S,
                    float* __restrict__ out, bf16* __restrict__ outbf) {
  const size_t row = blockIdx.x;
  const float* sr = S + row * TDIM;
  const int tid = threadIdx.x;
  float a = sr[tid], b = sr[tid + 256];
  __shared__ float red[256];
  red[tid] = fmaxf(a, b); __syncthreads();
  for (int o = 128; o > 0; o >>= 1) { if (tid < o) red[tid] = fmaxf(red[tid], red[tid + o]); __syncthreads(); }
  float mx = red[0];
  __syncthreads();
  float ea = __expf(a - mx), eb = __expf(b - mx);
  red[tid] = ea + eb; __syncthreads();
  for (int o = 128; o > 0; o >>= 1) { if (tid < o) red[tid] += red[tid + o]; __syncthreads(); }
  float inv = 1.f / red[0];
  ea *= inv; eb *= inv;
  out[row * TDIM + tid]        = ea;
  out[row * TDIM + tid + 256]  = eb;
  outbf[row * TDIM + tid]       = f2bf(ea);
  outbf[row * TDIM + tid + 256] = f2bf(eb);
}

// ------------------------------- host --------------------------------------
extern "C" void kernel_launch(void* const* d_in, const int* in_sizes, int n_in,
                              void* d_out, int out_size, void* d_ws, size_t ws_size,
                              hipStream_t stream) {
  (void)in_sizes; (void)n_in; (void)out_size; (void)ws_size;
  const int* X = (const int*)d_in[0];
  const int* Y = (const int*)d_in[1];
  auto P = [&](int i) { return (const float*)d_in[i]; };

  char* wsp = (char*)d_ws;
  auto alloc = [&](size_t bytes) -> void* {
    void* p = wsp; wsp += (bytes + 255) & ~(size_t)255; return p;
  };

  auto WT = [&](const float* W, int Kd, int Nd) -> bf16* {
    bf16* dst = (bf16*)alloc((size_t)Kd * Nd * sizeof(bf16));
    size_t total = (size_t)Kd * Nd;
    wt_kernel<<<dim3((unsigned)((total + 255) / 256)), dim3(256), 0, stream>>>(W, dst, Kd, Nd);
    return dst;
  };

  // --- weight conversion (bf16, transposed to [N,K]) ---
  bf16 *encWq[4], *encWo[4], *encW1[4], *encW2[4];
  bf16 *saWq[4], *saWo[4], *caWq[4], *caWo[4], *dW1[4], *dW2[4];
  const float *encBq[4], *encBo[4], *encB1[4], *encB2[4];
  const float *encG1[4], *encBt1[4], *encG2[4], *encBt2[4];
  const float *saBq[4], *saBo[4], *caBq[4], *caBo[4], *dB1[4], *dB2[4];
  const float *dG1[4], *dBt1[4], *dG2[4], *dBt2[4], *dG3[4], *dBt3[4];
  for (int i = 0; i < 4; ++i) {
    int e = 6 + 30 * i;
    encWq[i] = WT(P(e + 0), DDIM, DDIM);  encBq[i] = P(e + 1);
    encWo[i] = WT(P(e + 2), DDIM, DDIM);  encBo[i] = P(e + 3);
    encG1[i] = P(e + 4);  encBt1[i] = P(e + 5);
    encW1[i] = WT(P(e + 6), DDIM, FFDIM); encB1[i] = P(e + 7);
    encW2[i] = WT(P(e + 8), FFDIM, DDIM); encB2[i] = P(e + 9);
    encG2[i] = P(e + 10); encBt2[i] = P(e + 11);
    saWq[i] = WT(P(e + 12), DDIM, DDIM);  saBq[i] = P(e + 13);
    saWo[i] = WT(P(e + 14), DDIM, DDIM);  saBo[i] = P(e + 15);
    dG1[i] = P(e + 16); dBt1[i] = P(e + 17);
    caWq[i] = WT(P(e + 18), DDIM, DDIM);  caBq[i] = P(e + 19);
    caWo[i] = WT(P(e + 20), DDIM, DDIM);  caBo[i] = P(e + 21);
    dG2[i] = P(e + 22); dBt2[i] = P(e + 23);
    dW1[i] = WT(P(e + 24), DDIM, FFDIM);  dB1[i] = P(e + 25);
    dW2[i] = WT(P(e + 26), FFDIM, DDIM);  dB2[i] = P(e + 27);
    dG3[i] = P(e + 28); dBt3[i] = P(e + 29);
  }
  bf16* fcW = WT(P(126), DDIM, VY);
  const float* fcB = P(127);

  // --- activation buffers ---
  float* xbuf   = (float*)alloc((size_t)NTOK * DDIM * 4);
  bf16*  xbf    = (bf16*) alloc((size_t)NTOK * DDIM * 2);
  float* hbuf   = (float*)alloc((size_t)NTOK * DDIM * 4);
  bf16*  hbf    = (bf16*) alloc((size_t)NTOK * DDIM * 2);
  float* qbuf   = (float*)alloc((size_t)NTOK * DDIM * 4);
  float* kbuf   = (float*)alloc((size_t)NTOK * DDIM * 4);
  float* obuf   = (float*)alloc((size_t)NTOK * DDIM * 4);
  float* aout   = (float*)alloc((size_t)NTOK * DDIM * 4);
  bf16*  aoutbf = (bf16*) alloc((size_t)NTOK * DDIM * 2);
  bf16*  qh     = (bf16*) alloc((size_t)NTOK * DDIM * 2);
  bf16*  kh     = (bf16*) alloc((size_t)NTOK * DDIM * 2);
  bf16*  vT     = (bf16*) alloc((size_t)NTOK * DDIM * 2);
  float* Sbuf   = (float*)alloc((size_t)BB * HHEADS * TDIM * TDIM * 4);
  bf16*  Abf    = (bf16*) alloc((size_t)BB * HHEADS * TDIM * TDIM * 2);
  float* ffb    = (float*)alloc((size_t)NTOK * FFDIM * 4);
  bf16*  ffbf   = (bf16*) alloc((size_t)NTOK * FFDIM * 2);

  const int ZBH = BB * HHEADS;            // 128
  const unsigned EDG = (NTOK * DDIM) / 256;

  auto gemm_bias = [&](const bf16* A, const bf16* Bt, const float* bias,
                       float* C, bf16* Cbf, int M, int N, int K, bool relu) {
    dim3 g(M / 128, N / 128, 1);
    if (relu)
      gemm_kernel<128, 128, 32, 64, 0, true ><<<g, 256, 0, stream>>>(
          A, K, 0, Bt, K, 0, C, N, bias, Cbf, 1.f, 0, K);
    else
      gemm_kernel<128, 128, 32, 64, 0, false><<<g, 256, 0, stream>>>(
          A, K, 0, Bt, K, 0, C, N, bias, Cbf, 1.f, 0, K);
  };
  auto gemm_scores = [&](const bf16* Q, const bf16* Kh, float* S, int causal) {
    dim3 g(TDIM / 128, TDIM / 128, ZBH);
    gemm_kernel<128, 128, 32, 64, 1, false><<<g, 256, 0, stream>>>(
        Q, HD, (long long)TDIM * HD, Kh, HD, (long long)TDIM * HD,
        S, TDIM, nullptr, nullptr, 0.03125f /*1/sqrt(D)*/, causal, HD);
  };
  auto gemm_av = [&](const bf16* Att, const bf16* Vt, float* Out, bf16* Outbf) {
    dim3 g(TDIM / 128, HD / 64, ZBH);
    gemm_kernel<128, 64, 32, 32, 2, false><<<g, 256, 0, stream>>>(
        Att, TDIM, (long long)TDIM * TDIM, Vt, TDIM, (long long)HD * TDIM,
        Out, DDIM, nullptr, Outbf, 1.f, 0, TDIM);
  };
  auto softmax = [&](float* S, float* dstF, bf16* dstB) {
    softmax_kernel<<<dim3((unsigned)(ZBH * TDIM)), 256, 0, stream>>>(S, dstF, dstB);
  };
  auto lnorm = [&](const float* x, const float* res, const float* g,
                   const float* b2, float* out, bf16* outbf) {
    ln_kernel<<<dim3(NTOK), 256, 0, stream>>>(x, res, g, b2, out, outbf);
  };

  // ----------------------------- encoder -----------------------------------
  embed_kernel<<<dim3(EDG), 256, 0, stream>>>(X, P(2), P(3), xbuf, xbf);
  for (int i = 0; i < 4; ++i) {
    // bug-faithful: q = k = v = x @ Wq + bq
    gemm_bias(xbf, encWq[i], encBq[i], qbuf, nullptr, NTOK, DDIM, DDIM, false);
    headsplit_kernel<<<dim3(EDG), 256, 0, stream>>>(qbuf, qh, vT);
    gemm_scores(qh, qh, Sbuf, 0);
    softmax(Sbuf, Sbuf, Abf);
    gemm_av(Abf, vT, aout, aoutbf);
    gemm_bias(aoutbf, encWo[i], encBo[i], obuf, nullptr, NTOK, DDIM, DDIM, false);
    lnorm(xbuf, obuf, encG1[i], encBt1[i], xbuf, xbf);
    gemm_bias(xbf,  encW1[i], encB1[i], ffb,  ffbf,    NTOK, FFDIM, DDIM,  true);
    gemm_bias(ffbf, encW2[i], encB2[i], obuf, nullptr, NTOK, DDIM,  FFDIM, false);
    lnorm(xbuf, obuf, encG2[i], encBt2[i], xbuf, xbf);
  }
  // ctx = xbuf / xbf (left untouched by the decoder)

  // ----------------------------- decoder -----------------------------------
  float* outF  = (float*)d_out;
  float* saOut = outF  + (size_t)BB * TDIM * VY;
  float* caOut = saOut + (size_t)BB * HHEADS * TDIM * TDIM;

  embed_kernel<<<dim3(EDG), 256, 0, stream>>>(Y, P(4), P(5), hbuf, hbf);
  for (int i = 0; i < 4; ++i) {
    // masked self-attention (q = k = v from h)
    gemm_bias(hbf, saWq[i], saBq[i], qbuf, nullptr, NTOK, DDIM, DDIM, false);
    headsplit_kernel<<<dim3(EDG), 256, 0, stream>>>(qbuf, qh, vT);
    gemm_scores(qh, qh, Sbuf, 1);
    softmax(Sbuf, (i == 3) ? saOut : Sbuf, Abf);
    gemm_av(Abf, vT, aout, aoutbf);
    gemm_bias(aoutbf, saWo[i], saBo[i], obuf, nullptr, NTOK, DDIM, DDIM, false);
    lnorm(hbuf, obuf, dG1[i], dBt1[i], hbuf, hbf);
    // cross-attention (q from h; k = v = ctx @ Wq_ca + bq_ca)
    gemm_bias(hbf, caWq[i], caBq[i], qbuf, nullptr, NTOK, DDIM, DDIM, false);
    headsplit_kernel<<<dim3(EDG), 256, 0, stream>>>(qbuf, qh, nullptr);
    gemm_bias(xbf, caWq[i], caBq[i], kbuf, nullptr, NTOK, DDIM, DDIM, false);
    headsplit_kernel<<<dim3(EDG), 256, 0, stream>>>(kbuf, kh, vT);
    gemm_scores(qh, kh, Sbuf, 0);
    softmax(Sbuf, (i == 3) ? caOut : Sbuf, Abf);
    gemm_av(Abf, vT, aout, aoutbf);
    gemm_bias(aoutbf, caWo[i], caBo[i], obuf, nullptr, NTOK, DDIM, DDIM, false);
    lnorm(hbuf, obuf, dG2[i], dBt2[i], hbuf, hbf);
    // FFN; bug-faithful: LN of FFN output alone (no residual)
    gemm_bias(hbf,  dW1[i], dB1[i], ffb,  ffbf,    NTOK, FFDIM, DDIM,  true);
    gemm_bias(ffbf, dW2[i], dB2[i], obuf, nullptr, NTOK, DDIM,  FFDIM, false);
    lnorm(obuf, nullptr, dG3[i], dBt3[i], hbuf, hbf);
  }

  // logits = h @ fc_W + fc_b
  gemm_bias(hbf, fcW, fcB, outF, nullptr, NTOK, VY, DDIM, false);
}
